// LinearAttention_80075370266840
// MI455X (gfx1250) — matrix operand order repaired and checked by reading, not compiled, not run on hardware
//
#include <hip/hip_runtime.h>

// ---------------------------------------------------------------------------
// Problem constants (match reference)
// ---------------------------------------------------------------------------
#define BB      4
#define TT      4096
#define DIMM    1024
#define HEADS   16
#define DH      64
#define INNER   1024          // HEADS * DH
#define N3      3072          // 3 * INNER
#define MTOK    (BB * TT)     // 16384 tokens
#define SCALEQ  0.125f        // DH^-0.5

// ---------------------------------------------------------------------------
// bf16 / WMMA plumbing (gfx1250, wave32)
// ---------------------------------------------------------------------------
typedef __attribute__((ext_vector_type(16))) __bf16 v16bf;
typedef __attribute__((ext_vector_type(8)))  float  v8f;
typedef __attribute__((ext_vector_type(4)))  unsigned int uint4v;
typedef __attribute__((ext_vector_type(8)))  int     int8v;
typedef __attribute__((ext_vector_type(4)))  int     int4v;

union FragU { v16bf v; unsigned int u[8]; };

__device__ __forceinline__ unsigned short f2bf(float f) {
    union { float f; unsigned int u; } c; c.f = f;
    unsigned int u = c.u;
    unsigned int r = (u + 0x7FFFu + ((u >> 16) & 1u)) >> 16;   // round-nearest-even
    return (unsigned short)r;
}
__device__ __forceinline__ float bf2f(unsigned short b) {
    union { unsigned int u; float f; } c; c.u = ((unsigned int)b) << 16;
    return c.f;
}
// pack two f32 -> two bf16 (truncate) in ONE v_perm_b32
__device__ __forceinline__ unsigned int f2bf_pk(float lo, float hi) {
    return __builtin_amdgcn_perm(__float_as_uint(hi), __float_as_uint(lo), 0x07060302u);
}
// interleave hi16/lo16 of two dwords (for bf16 transpose packing)
__device__ __forceinline__ unsigned int pk_lo16(unsigned int a, unsigned int b) {
    return __builtin_amdgcn_perm(b, a, 0x05040100u);   // {a[15:0], b[15:0]}
}
__device__ __forceinline__ unsigned int pk_hi16(unsigned int a, unsigned int b) {
    return __builtin_amdgcn_perm(b, a, 0x07060302u);   // {a[31:16], b[31:16]}
}

// A fragment: tile stored [16(m)][ldk(k)] bf16, ldk even.
// A 16x32: lane = half*16+m ; VGPR v: K=half*16+2v (same k order as ld_frag_b)
__device__ __forceinline__ v16bf ld_frag_a(const unsigned short* base, int ldk) {
    int lane = threadIdx.x & 31;
    int m = lane & 15, half = lane >> 4;
    const unsigned int* row = (const unsigned int*)(base + m * ldk);
    FragU f;
#pragma unroll
    for (int v = 0; v < 8; ++v) f.u[v] = row[half * 8 + v];
    return f.v;
}

// B fragment: B stored transposed in LDS as [16(n)][ldk(k)] bf16.
// B k order used here (same as ld_frag_a): lanes 0-15 K=0..15, lanes 16-31 K=16..31; VGPR v: K=half*16+2v
__device__ __forceinline__ v16bf ld_frag_b(const unsigned short* base, int ldk) {
    int lane = threadIdx.x & 31;
    int n = lane & 15, half = lane >> 4;
    const unsigned int* row = (const unsigned int*)(base + n * ldk);
    FragU f;
#pragma unroll
    for (int v = 0; v < 8; ++v) f.u[v] = row[half * 8 + v];
    return f.v;
}

__device__ __forceinline__ v8f wmma_bf16(v16bf a, v16bf b, v8f c) {
    return __builtin_amdgcn_wmma_f32_16x16x32_bf16(false, a, false, b, (short)0, c, false, false);
}

// ---------------------------------------------------------------------------
// Tensor Data Mover: 2D bf16 tile load global -> LDS with LDS row padding.
// D# per CDNA5 ISA ch.8 (group0: count/lds/global/type, group1: sizes/strides).
// This toolchain exposes the 6-arg builtin:
//   (uint32x4 g0, int32x8 g1, int32x4 g2, int32x4 g3, int32x8 aux, i32 cpol)
// ---------------------------------------------------------------------------
#if __has_builtin(__builtin_amdgcn_tensor_load_to_lds)
#define HAVE_TDM 1
#else
#define HAVE_TDM 0
#endif

#if HAVE_TDM
__device__ __forceinline__ void tdm_load_2d_bf16(
    unsigned int lds_byte_addr, const void* gptr,
    unsigned int tensor_d0, unsigned int tensor_d1,
    unsigned int tile_d0, unsigned int tile_d1,
    unsigned long long stride0_elems,
    unsigned int pad_interval_code, unsigned int pad_amount_code)
{
    unsigned long long ga = (unsigned long long)(size_t)gptr;
    uint4v g0;
    g0[0] = 1u;                                          // count=1, user mode
    g0[1] = lds_byte_addr;                               // lds_addr
    g0[2] = (unsigned int)(ga & 0xFFFFFFFFu);            // global_addr[31:0]
    g0[3] = (unsigned int)((ga >> 32) & 0x01FFFFFFu) | 0x80000000u; // addr[56:32] | type=2
    int8v g1;
    unsigned int w0 = (1u << 16)                         // data_size = 1 (2 bytes)
                    | (1u << 20)                         // pad_enable
                    | (pad_interval_code << 22)
                    | (pad_amount_code   << 25);
    g1[0] = (int)w0;
    g1[1] = (int)((tensor_d0 & 0xFFFFu) << 16);                          // barrier_addr=0 | dim0 lo
    g1[2] = (int)((tensor_d0 >> 16) | ((tensor_d1 & 0xFFFFu) << 16));    // dim0 hi | dim1 lo
    g1[3] = (int)((tensor_d1 >> 16) | (tile_d0 << 16));                  // dim1 hi | tile_dim0
    g1[4] = (int)(tile_d1 & 0xFFFFu);                                    // tile_dim1 | tile_dim2=0
    g1[5] = (int)(stride0_elems & 0xFFFFFFFFull);                        // dim0_stride lo
    g1[6] = (int)((stride0_elems >> 32) & 0xFFFFull);                    // dim0_stride hi | dim1_stride=0
    g1[7] = 0;
    int4v z4 = {0, 0, 0, 0};
    int8v z8 = {0, 0, 0, 0, 0, 0, 0, 0};
    __builtin_amdgcn_tensor_load_to_lds(g0, g1, z4, z4, z8, 0);
}
#endif

// ---------------------------------------------------------------------------
// K1: qkv = x @ W_qkv  (M=16384, K=1024, N=3072), f32 in -> bf16 q/k/v out
// ---------------------------------------------------------------------------
#define LDK1 40
__global__ __launch_bounds__(256) void qkv_gemm_kernel(
    const float* __restrict__ x, const float* __restrict__ Wqkv,
    unsigned short* __restrict__ qb, unsigned short* __restrict__ kb,
    unsigned short* __restrict__ vb)
{
    __shared__ unsigned short As[128 * LDK1];   // [m][k]
    __shared__ unsigned short Bs[64 * LDK1];    // [n][k] (B transposed)

    const int tid  = threadIdx.x;
    const int wave = tid >> 5;
    const int lane = tid & 31;
    const int wm = wave & 3, wn = wave >> 2;
    const int m0 = blockIdx.y * 128;
    const int n0 = blockIdx.x * 64;

    v8f acc[2][2];
    const v8f vz = {0.f,0.f,0.f,0.f,0.f,0.f,0.f,0.f};
#pragma unroll
    for (int i = 0; i < 2; ++i)
#pragma unroll
        for (int j = 0; j < 2; ++j) acc[i][j] = vz;

    // staging coordinates
    const int arow = tid >> 1, acb = (tid & 1) * 16;           // A: 16 f32 each
    const int bnb  = (tid & 15) * 4, bk2 = (tid >> 4) * 2;     // B: 2 k-rows x 4 n

    for (int kk = 0; kk < DIMM; kk += 32) {
        // stage A: x[m0+arow][kk..+31] -> bf16 packed dwords
        {
            const float* src = x + (size_t)(m0 + arow) * DIMM + kk + acb;
            unsigned int* dst = (unsigned int*)(As + arow * LDK1 + acb);
#pragma unroll
            for (int j = 0; j < 8; ++j) dst[j] = f2bf_pk(src[2*j], src[2*j+1]);
            if (kk + 32 < DIMM) __builtin_prefetch(src + 32, 0, 3);
        }
        // stage B transposed: pack (k, k+1) pairs -> Bs[n][k] dwords
        {
            const float* s0 = Wqkv + (size_t)(kk + bk2)     * N3 + n0 + bnb;
            const float* s1 = Wqkv + (size_t)(kk + bk2 + 1) * N3 + n0 + bnb;
#pragma unroll
            for (int j = 0; j < 4; ++j) {
                unsigned int p = f2bf_pk(s0[j], s1[j]);
                *(unsigned int*)(Bs + (bnb + j) * LDK1 + bk2) = p;
            }
            if (kk + 32 < DIMM) __builtin_prefetch(s0 + 2 * N3, 0, 3);
        }
        __syncthreads();

        v16bf af[2], bf[2];
#pragma unroll
        for (int mt = 0; mt < 2; ++mt) af[mt] = ld_frag_a(As + (wm * 32 + mt * 16) * LDK1, LDK1);
#pragma unroll
        for (int nt = 0; nt < 2; ++nt) bf[nt] = ld_frag_b(Bs + (wn * 32 + nt * 16) * LDK1, LDK1);
#pragma unroll
        for (int mt = 0; mt < 2; ++mt)
#pragma unroll
            for (int nt = 0; nt < 2; ++nt) acc[mt][nt] = wmma_bf16(af[mt], bf[nt], acc[mt][nt]);
        __syncthreads();
    }

    // epilogue: region (q/k/v) is BLOCK-uniform since 64 | 1024
    const int region = n0 >> 10;           // 0=q 1=k 2=v
    const int col0   = n0 & 1023;
    unsigned short* dstbuf = (region == 0) ? qb : ((region == 1) ? kb : vb);
    const float scl = (region == 0) ? SCALEQ : 1.0f;

    const int half = lane >> 4, ncol = lane & 15;
#pragma unroll
    for (int mt = 0; mt < 2; ++mt) {
#pragma unroll
        for (int nt = 0; nt < 2; ++nt) {
            int gn = col0 + wn * 32 + nt * 16 + ncol;
#pragma unroll
            for (int r = 0; r < 8; ++r) {
                int gm = m0 + wm * 32 + mt * 16 + r + 8 * half;
                dstbuf[(size_t)gm * INNER + gn] = f2bf(acc[mt][nt][r] * scl);
            }
        }
    }
}

// ---------------------------------------------------------------------------
// K2: softmax of k over sequence dim T (per b, channel), then zero padded
//     positions AFTER softmax (faithful to reference). In-place bf16.
// ---------------------------------------------------------------------------
__global__ __launch_bounds__(256) void k_softmax_kernel(
    unsigned short* __restrict__ kb, const unsigned char* __restrict__ mask)
{
    const int c = blockIdx.x * 256 + threadIdx.x;   // channel 0..1023
    const int b = blockIdx.y;
    unsigned short* base = kb + (size_t)b * TT * INNER + c;
    const unsigned char* mrow = mask + (size_t)b * TT;

    float mx = -3.4e38f;
    for (int t = 0; t < TT; ++t) mx = fmaxf(mx, bf2f(base[(size_t)t * INNER]));
    float s = 0.f;
    for (int t = 0; t < TT; ++t) s += __expf(bf2f(base[(size_t)t * INNER]) - mx);
    float inv = 1.f / s;
    for (int t = 0; t < TT; ++t) {
        float e = mrow[t] ? 0.f : __expf(bf2f(base[(size_t)t * INNER]) - mx) * inv;
        base[(size_t)t * INNER] = f2bf(e);
    }
}

// ---------------------------------------------------------------------------
// K3: context partials: ctx_part[s][bh][64][64] = sum_{t in slice} k^T v
//     grid (64 bh, 8 slices), block 128 thr = 4 waves
// ---------------------------------------------------------------------------
#define LDK3 40
__global__ __launch_bounds__(128) void context_gemm_kernel(
    const unsigned short* __restrict__ kb, const unsigned short* __restrict__ vb,
    float* __restrict__ ctx_part)
{
    __shared__ unsigned short Ks[64 * LDK3];  // [d][t]
    __shared__ unsigned short Vs[64 * LDK3];  // [e][t]

    const int tid  = threadIdx.x;
    const int wave = tid >> 5;
    const int lane = tid & 31;
    const int bh = blockIdx.x;
    const int b = bh >> 4, h = bh & 15;
    const int ch0 = h * DH;
    const int slice = blockIdx.y;

    v8f acc[4];
    const v8f vz = {0.f,0.f,0.f,0.f,0.f,0.f,0.f,0.f};
#pragma unroll
    for (int i = 0; i < 4; ++i) acc[i] = vz;

    // staging: 2 consecutive t-rows x 8 channels per thread, pack along t
    const int c8 = (tid & 7) * 8;
    const int tp = (tid >> 3) * 2;   // 0..30

    const int tbeg = slice * (TT / 8);
    for (int t0 = tbeg; t0 < tbeg + TT / 8; t0 += 32) {
        const uint4v* k0 = (const uint4v*)(kb + (size_t)(b * TT + t0 + tp)     * INNER + ch0 + c8);
        const uint4v* k1 = (const uint4v*)(kb + (size_t)(b * TT + t0 + tp + 1) * INNER + ch0 + c8);
        const uint4v* v0 = (const uint4v*)(vb + (size_t)(b * TT + t0 + tp)     * INNER + ch0 + c8);
        const uint4v* v1 = (const uint4v*)(vb + (size_t)(b * TT + t0 + tp + 1) * INNER + ch0 + c8);
        uint4v ka = *k0, kbv = *k1, va = *v0, vbv = *v1;
#pragma unroll
        for (int j = 0; j < 4; ++j) {
            *(unsigned int*)(Ks + (c8 + 2*j    ) * LDK3 + tp) = pk_lo16(ka[j], kbv[j]);
            *(unsigned int*)(Ks + (c8 + 2*j + 1) * LDK3 + tp) = pk_hi16(ka[j], kbv[j]);
            *(unsigned int*)(Vs + (c8 + 2*j    ) * LDK3 + tp) = pk_lo16(va[j], vbv[j]);
            *(unsigned int*)(Vs + (c8 + 2*j + 1) * LDK3 + tp) = pk_hi16(va[j], vbv[j]);
        }
        __syncthreads();

        v16bf af = ld_frag_a(Ks + (wave * 16) * LDK3, LDK3);
#pragma unroll
        for (int nt = 0; nt < 4; ++nt) {
            v16bf bfv = ld_frag_b(Vs + (nt * 16) * LDK3, LDK3);
            acc[nt] = wmma_bf16(af, bfv, acc[nt]);
        }
        __syncthreads();
    }

    const int half = lane >> 4, ncol = lane & 15;
    float* dst = ctx_part + ((size_t)slice * 64 + bh) * (64 * 64);
#pragma unroll
    for (int nt = 0; nt < 4; ++nt) {
        int e = nt * 16 + ncol;
#pragma unroll
        for (int r = 0; r < 8; ++r) {
            int d = wave * 16 + r + 8 * half;
            dst[d * 64 + e] = acc[nt][r];
        }
    }
}

// ---------------------------------------------------------------------------
// K4: attn = q @ context per (b,h): M=T, K=64, N=64.
//     grid (T/128, 64 bh), block 256 thr = 8 waves.
//     Q tile loaded via Tensor Data Mover (one 16-row slab per wave).
// ---------------------------------------------------------------------------
#define LDK4 72
__global__ __launch_bounds__(256) void q_ctx_gemm_kernel(
    const unsigned short* __restrict__ qb, const float* __restrict__ ctx_part,
    unsigned short* __restrict__ attn)
{
    __shared__ unsigned short Qs[128 * LDK4];  // [m][k]
    __shared__ unsigned short Cs[64 * LDK4];   // [e][d]

    const int tid  = threadIdx.x;
    const int wave = tid >> 5;
    const int lane = tid & 31;
    const int bh = blockIdx.y;
    const int b = bh >> 4, h = bh & 15;
    const int ch0 = h * DH;
    const int t0 = blockIdx.x * 128;

    // ---- stage Q (pure bf16 tile copy) ----
#if HAVE_TDM
    {
        // wave id as a provably wave-uniform scalar
        int w = __builtin_amdgcn_readfirstlane(tid) >> 5;
        // 16 rows of 64 bf16; LDS row = 128B (32 dw) + 16B pad -> interval code 4, amount code 3
        tdm_load_2d_bf16((unsigned int)(size_t)Qs + (unsigned int)(w * 16 * LDK4 * 2),
                         qb + (size_t)(b * TT + t0 + w * 16) * INNER + ch0,
                         /*tensor_d0*/ INNER, /*tensor_d1*/ (unsigned)MTOK,
                         /*tile_d0*/ 64, /*tile_d1*/ 16,
                         /*stride0*/ INNER,
                         /*pad_interval*/ 4, /*pad_amount*/ 3);
        __builtin_amdgcn_s_wait_tensorcnt(0);
    }
#else
    {
        int row = tid >> 1;
        int cb  = (tid & 1) * 32;
        const unsigned short* src = qb + (size_t)(b * TT + t0 + row) * INNER + ch0 + cb;
        unsigned short* dst = Qs + row * LDK4 + cb;
#pragma unroll
        for (int j = 0; j < 32; ++j) dst[j] = src[j];
    }
#endif
    // ---- stage context: sum 8 partials, transpose -> Cs[e][d], packed dwords ----
    {
        int e    = tid & 63;
        int dblk = (tid >> 6) * 16;
        float s[16];
#pragma unroll
        for (int j = 0; j < 16; ++j) {
            float acc_s = 0.f;
#pragma unroll
            for (int p = 0; p < 8; ++p)
                acc_s += ctx_part[((size_t)p * 64 + bh) * (64 * 64) + (dblk + j) * 64 + e];
            s[j] = acc_s;
        }
        unsigned int* dst = (unsigned int*)(Cs + e * LDK4 + dblk);
#pragma unroll
        for (int j = 0; j < 8; ++j) dst[j] = f2bf_pk(s[2*j], s[2*j+1]);
    }
    __syncthreads();

    v8f acc[4];
    const v8f vz = {0.f,0.f,0.f,0.f,0.f,0.f,0.f,0.f};
#pragma unroll
    for (int i = 0; i < 4; ++i) acc[i] = vz;

#pragma unroll
    for (int kk = 0; kk < 64; kk += 32) {
        v16bf af = ld_frag_a(Qs + (wave * 16) * LDK4 + kk, LDK4);
#pragma unroll
        for (int nt = 0; nt < 4; ++nt) {
            v16bf bfv = ld_frag_b(Cs + (nt * 16) * LDK4 + kk, LDK4);
            acc[nt] = wmma_bf16(af, bfv, acc[nt]);
        }
    }

    const int half = lane >> 4, ncol = lane & 15;
#pragma unroll
    for (int nt = 0; nt < 4; ++nt) {
        int gn = nt * 16 + ncol;
#pragma unroll
        for (int r = 0; r < 8; ++r) {
            int gm = t0 + wave * 16 + r + 8 * half;
            attn[(size_t)(b * TT + gm) * INNER + ch0 + gn] = f2bf(acc[nt][r]);
        }
    }
}

// ---------------------------------------------------------------------------
// K5: out = attn @ W_out + b_out  (M=16384, K=1024, N=1024) -> f32
//     A tile loaded via Tensor Data Mover (one 16-row slab per wave).
// ---------------------------------------------------------------------------
__global__ __launch_bounds__(256) void out_gemm_kernel(
    const unsigned short* __restrict__ attn, const float* __restrict__ Wout,
    const float* __restrict__ bias, float* __restrict__ out)
{
    __shared__ unsigned short As[128 * LDK1];
    __shared__ unsigned short Bs[64 * LDK1];

    const int tid  = threadIdx.x;
    const int wave = tid >> 5;
    const int lane = tid & 31;
    const int wm = wave & 3, wn = wave >> 2;
    const int m0 = blockIdx.y * 128;
    const int n0 = blockIdx.x * 64;
#if HAVE_TDM
    const int wuni = __builtin_amdgcn_readfirstlane(tid) >> 5;   // uniform wave id
#endif

    v8f acc[2][2];
    const v8f vz = {0.f,0.f,0.f,0.f,0.f,0.f,0.f,0.f};
#pragma unroll
    for (int i = 0; i < 2; ++i)
#pragma unroll
        for (int j = 0; j < 2; ++j) acc[i][j] = vz;

    const int bnb = (tid & 15) * 4, bk2 = (tid >> 4) * 2;

    for (int kk = 0; kk < INNER; kk += 32) {
        // stage A via TDM: each wave DMAs its 16-row slab (32 bf16/row, 64B + 16B pad)
#if HAVE_TDM
        tdm_load_2d_bf16((unsigned int)(size_t)As + (unsigned int)(wuni * 16 * LDK1 * 2),
                         attn + (size_t)(m0 + wuni * 16) * INNER + kk,
                         /*tensor_d0*/ INNER, /*tensor_d1*/ (unsigned)MTOK,
                         /*tile_d0*/ 32, /*tile_d1*/ 16,
                         /*stride0*/ INNER,
                         /*pad_interval*/ 3, /*pad_amount*/ 3);
#else
        {
            int row = tid >> 1;
            int cb  = (tid & 1) * 16;
            const unsigned short* src = attn + (size_t)(m0 + row) * INNER + kk + cb;
            unsigned short* dst = As + row * LDK1 + cb;
#pragma unroll
            for (int j = 0; j < 16; ++j) dst[j] = src[j];
        }
#endif
        // stage B: W_out f32 -> bf16 packed, transposed
        {
            const float* s0 = Wout + (size_t)(kk + bk2)     * DIMM + n0 + bnb;
            const float* s1 = Wout + (size_t)(kk + bk2 + 1) * DIMM + n0 + bnb;
#pragma unroll
            for (int j = 0; j < 4; ++j) {
                unsigned int p = f2bf_pk(s0[j], s1[j]);
                *(unsigned int*)(Bs + (bnb + j) * LDK1 + bk2) = p;
            }
            if (kk + 32 < INNER) __builtin_prefetch(s0 + 2 * DIMM, 0, 3);
        }
#if HAVE_TDM
        __builtin_amdgcn_s_wait_tensorcnt(0);
#endif
        __syncthreads();

        v16bf af[2], bf[2];
#pragma unroll
        for (int mt = 0; mt < 2; ++mt) af[mt] = ld_frag_a(As + (wm * 32 + mt * 16) * LDK1, LDK1);
#pragma unroll
        for (int nt = 0; nt < 2; ++nt) bf[nt] = ld_frag_b(Bs + (wn * 32 + nt * 16) * LDK1, LDK1);
#pragma unroll
        for (int mt = 0; mt < 2; ++mt)
#pragma unroll
            for (int nt = 0; nt < 2; ++nt) acc[mt][nt] = wmma_bf16(af[mt], bf[nt], acc[mt][nt]);
        __syncthreads();
    }

    const int half = lane >> 4, ncol = lane & 15;
#pragma unroll
    for (int mt = 0; mt < 2; ++mt) {
#pragma unroll
        for (int nt = 0; nt < 2; ++nt) {
            int gn = n0 + wn * 32 + nt * 16 + ncol;
            float bv = bias[gn];
#pragma unroll
            for (int r = 0; r < 8; ++r) {
                int gm = m0 + wm * 32 + mt * 16 + r + 8 * half;
                out[(size_t)gm * DIMM + gn] = acc[mt][nt][r] + bv;
            }
        }
    }
}

// ---------------------------------------------------------------------------
// Launch
// ---------------------------------------------------------------------------
extern "C" void kernel_launch(void* const* d_in, const int* in_sizes, int n_in,
                              void* d_out, int out_size, void* d_ws, size_t ws_size,
                              hipStream_t stream)
{
    (void)in_sizes; (void)n_in; (void)out_size; (void)ws_size;

    const float*         x     = (const float*)d_in[0];
    const unsigned char* mask  = (const unsigned char*)d_in[1];   // bool array
    const float*         Wqkv  = (const float*)d_in[2];
    const float*         Wout  = (const float*)d_in[3];
    const float*         bout  = (const float*)d_in[4];
    float*               out   = (float*)d_out;

    // workspace layout (bytes)
    char* ws = (char*)d_ws;
    const size_t SZ_BF = (size_t)MTOK * INNER * sizeof(unsigned short);  // 32 MB
    unsigned short* qb   = (unsigned short*)(ws);
    unsigned short* kb   = (unsigned short*)(ws + SZ_BF);
    unsigned short* vb   = (unsigned short*)(ws + 2 * SZ_BF);
    unsigned short* attn = (unsigned short*)(ws + 3 * SZ_BF);
    float*          ctxp = (float*)(ws + 4 * SZ_BF);                     // 8*64*64*64 f32 = 8 MB

    qkv_gemm_kernel<<<dim3(N3 / 64, MTOK / 128), 256, 0, stream>>>(x, Wqkv, qb, kb, vb);
    k_softmax_kernel<<<dim3(INNER / 256, BB), 256, 0, stream>>>(kb, mask);
    context_gemm_kernel<<<dim3(BB * HEADS, 8), 128, 0, stream>>>(kb, vb, ctxp);
    q_ctx_gemm_kernel<<<dim3(TT / 128, BB * HEADS), 256, 0, stream>>>(qb, ctxp, attn);
    out_gemm_kernel<<<dim3(DIMM / 64, MTOK / 128), 256, 0, stream>>>(attn, Wout, bout, out);
}
